// EncodeProcessDecode_72945724555854
// MI455X (gfx1250) — compile-verified, matrix operand order checked by reading, not run on hardware
//
#include <hip/hip_runtime.h>
#include <hip/hip_bf16.h>

typedef __attribute__((ext_vector_type(16))) _Float16 v16h;
typedef __attribute__((ext_vector_type(8)))  float    v8f;
typedef int i4v __attribute__((vector_size(16)));

struct __attribute__((aligned(16))) H8 { _Float16 v[8]; };
struct __attribute__((aligned(8)))  H4 { _Float16 v[4]; };

#if defined(__has_builtin)
#if __has_builtin(__builtin_amdgcn_global_load_async_to_lds_b128)
#define HAVE_ASYNC_LDS 1
typedef __attribute__((address_space(1))) i4v* gp1;
typedef __attribute__((address_space(3))) i4v* lp3;
#endif
#endif

#define NVN 30000
#define NEN 300000
#define NUN 64

// ---------------- param leaf indices (jax tree-flatten, sorted dict keys) ----
enum {
  IN_X = 0, IN_EA = 1, IN_U = 2,
  P_A_DST = 3, P_A_EDGE = 4, P_A_SRC = 5, P_ATT_B = 6, P_ATT_W = 7, P_ATT_BIAS = 8,
  P_LIN = 9, P_LIN_E = 10, P_ATT_LNB = 11, P_ATT_LNG = 12,
  P_LSTMB_BHH = 13, P_LSTMB_BIH = 14, P_LSTMB_WHH = 15, P_LSTMB_WIH = 16,
  P_LSTMF_BHH = 17, P_LSTMF_BIH = 18, P_LSTMF_WHH = 19, P_LSTMF_WIH = 20,
  P_CE_B0 = 21, P_CE_B1 = 22, P_CE_LNB = 23, P_CE_LNG = 24, P_CE_W0 = 25, P_CE_W1 = 26,
  P_CU_B0 = 27, P_CU_B1 = 28, P_CU_LNB = 29, P_CU_LNG = 30, P_CU_W0 = 31, P_CU_W1 = 32,
  P_CV_B0 = 33, P_CV_B1 = 34, P_CV_LNB = 35, P_CV_LNG = 36, P_CV_W0 = 37, P_CV_W1 = 38,
  P_DE_B0 = 39, P_DE_B1 = 40, P_DE_LNB = 41, P_DE_LNG = 42, P_DE_W0 = 43, P_DE_W1 = 44,
  P_DU_B0 = 45, P_DU_B1 = 46, P_DU_LNB = 47, P_DU_LNG = 48, P_DU_W0 = 49, P_DU_W1 = 50,
  P_DV_B0 = 51, P_DV_B1 = 52, P_DV_LNB = 53, P_DV_LNG = 54, P_DV_W0 = 55, P_DV_W1 = 56,
  P_EE_B0 = 57, P_EE_B1 = 58, P_EE_LNB = 59, P_EE_LNG = 60, P_EE_W0 = 61, P_EE_W1 = 62,
  P_EU_B0 = 63, P_EU_B1 = 64, P_EU_LNB = 65, P_EU_LNG = 66, P_EU_W0 = 67, P_EU_W1 = 68,
  P_EV_B0 = 69, P_EV_B1 = 70, P_EV_LNB = 71, P_EV_LNG = 72, P_EV_W0 = 73, P_EV_W1 = 74,
  P_OE_B = 75, P_OE_W = 76, P_OU_B = 77, P_OU_W = 78, P_OV_B = 79, P_OV_W = 80,
  IN_EIDX = 81, IN_VIDX = 82, IN_EINDICES = 83
};

// ---------------- gather-fused GEMM A operand descriptor ----------------
// Segment boundaries are all multiples of 32 (or a single short segment), so a
// 32-wide K-chunk never straddles segments: selection is uniform per chunk.
struct GemmA {
  const float *base0, *base1, *base2, *base3;
  const int   *idx0,  *idx1,  *idx2,  *idx3;
  int kw0, kw1, kw2, kw3;     // k-width of each concat segment
  int st0, st1, st2, st3;     // row stride of each segment
};

// swizzled weight index: fragment-chunk layout so GEMM B-staging is a memcpy.
// Element (gk, n) of a [K,N] row-major weight -> chunk c=gk/32, tile nt=n/16,
// lane = ((gk>>3)&1)*16 + (n&15), fragpos = (gk&7) + ((gk>>4)<<3).
__device__ __forceinline__ size_t wswz(int gk, int n, int N) {
  int c    = gk >> 5, kk = gk & 31;
  int fp   = (kk & 7) + ((kk >> 4) << 3);
  int lane = (((kk >> 3) & 1) << 4) | (n & 15);
  return (((size_t)c * (N >> 4) + (n >> 4)) * 32 + lane) * 16 + fp;
}

// ---------------- WMMA GEMM: out[M,N] = relu(A[M,K] @ W[K,N] + bias) ----------------
// 256 threads = 8 waves, 32-row stripe per block (2 wave-groups of 16 rows share
// one staged W chunk). Wave quadrant wq owns N-tiles {wq, wq+4, ...}. A staging:
// one predicated float4 gather per thread per chunk (segment chosen uniformly
// from k0). B staging: GLOBAL_LOAD_ASYNC_TO_LDS_B128 of pre-swizzled weights
// (ASYNCcnt-tracked DMA, no VGPR round trip), with a synchronous b128-copy
// fallback. Fragments are plain 16B LDS loads; f16 operands, f32 accumulate via
// v_wmma_f32_16x16x32_f16.
template<int N>
__global__ __launch_bounds__(256)
void gemm_wmma_kernel(GemmA A, const _Float16* __restrict__ Wsw,
                      const float* __restrict__ bias, float* __restrict__ out,
                      long ldc, int M, int K, int relu)
{
  constexpr int NT  = N / 16;   // 4, 8, 16 column tiles
  constexpr int TPW = NT / 4;   // tiles per wave: 1, 2, 4

  __shared__ __align__(16) _Float16 As[32][40];   // 32 x 32 (+pad)
  __shared__ __align__(32) _Float16 Bs[NT * 512]; // fragment-ready layout

  const int tid  = threadIdx.x;
  const int lane = tid & 31;
  const int wid  = tid >> 5;
  const int wg   = wid >> 2;        // row group (0/1)
  const int wq   = wid & 3;         // tile quadrant
  const int half = lane >> 4;
  const int mr   = lane & 15;
  const int m0   = blockIdx.x * 32 + wg * 16;

  // A-staging role: this thread owns 4 consecutive k of one row
  const int ar = tid >> 3;          // row 0..31
  const int ak = (tid & 7) << 2;    // k-in-chunk 0..28
  const int am = blockIdx.x * 32 + ar;
  const int amv = (am < M);
  const int t0 = A.kw0, t1 = t0 + A.kw1, t2 = t1 + A.kw2;

  v8f acc[TPW];
  #pragma unroll
  for (int t = 0; t < TPW; ++t)
    #pragma unroll
    for (int j = 0; j < 8; ++j) acc[t][j] = 0.f;

  for (int k0 = 0; k0 < K; k0 += 32) {
    // ---- uniform (scalar) segment select for this chunk ----
    const float* sb = A.base0; const int* si = A.idx0; int st = A.st0; int ko = 0;
    if (k0 >= t2)      { sb = A.base3; si = A.idx3; st = A.st3; ko = t2; }
    else if (k0 >= t1) { sb = A.base2; si = A.idx2; st = A.st2; ko = t1; }
    else if (k0 >= t0) { sb = A.base1; si = A.idx1; st = A.st1; ko = t0; }

    // ---- stage B: async DMA (or b128 copy) of pre-swizzled chunk ----
    const _Float16* wsrc = Wsw + (size_t)(k0 >> 5) * (NT * 512);
#ifdef HAVE_ASYNC_LDS
    #pragma unroll
    for (int e = 0; e < (NT * 64) / 256; ++e) {
      int c = tid + e * 256;
      _Float16* gsrc = const_cast<_Float16*>(&wsrc[(size_t)c * 8]);
      __builtin_amdgcn_global_load_async_to_lds_b128(
          (gp1)gsrc, (lp3)&Bs[c * 8], 0, 0);
    }
#else
    #pragma unroll
    for (int e = 0; e < (NT * 64) / 256; ++e) {
      int c = tid + e * 256;
      *(H8*)&Bs[c * 8] = *(const H8*)&wsrc[(size_t)c * 8];
    }
#endif

    // ---- stage A: one predicated float4 gather -> 4 f16 -> 8B LDS store ----
    int gk = k0 + ak;
    float4 v = {0.f, 0.f, 0.f, 0.f};
    if (amv && gk < K) {
      int row = si ? si[am] : am;
      v = *(const float4*)&sb[(long)row * st + (gk - ko)];
    }
    H4 h;
    h.v[0] = (_Float16)v.x; h.v[1] = (_Float16)v.y;
    h.v[2] = (_Float16)v.z; h.v[3] = (_Float16)v.w;
    *(H4*)&As[ar][ak] = h;

#ifdef HAVE_ASYNC_LDS
    asm volatile("s_wait_asynccnt 0x0" ::: "memory");
#endif
    __syncthreads();

    // ---- A fragment: two aligned 16B LDS loads ----
    const int arow = wg * 16 + mr;
    H8 alo = *(const H8*)&As[arow][half * 8];
    H8 ahi = *(const H8*)&As[arow][half * 8 + 16];
    v16h a;
    #pragma unroll
    for (int j = 0; j < 8; ++j) { a[j] = alo.v[j]; a[8 + j] = ahi.v[j]; }

    // ---- B fragments: one aligned 32B LDS load each; back-to-back WMMA ----
    #pragma unroll
    for (int t = 0; t < TPW; ++t) {
      int nt = wq + 4 * t;
      v16h b = *(const v16h*)&Bs[(nt * 32 + lane) * 16];
      acc[t] = __builtin_amdgcn_wmma_f32_16x16x32_f16(false, a, false, b, (short)0, acc[t], false, false);
    }
    __syncthreads();
  }

  // ---- epilogue: C layout (VGPR r -> row r + 8*half, col = lane&15) ----
  #pragma unroll
  for (int t = 0; t < TPW; ++t) {
    int nt = wq + 4 * t;
    #pragma unroll
    for (int r = 0; r < 8; ++r) {
      int row = m0 + r + 8 * half;
      int col = nt * 16 + mr;
      if (row < M) {
        float v = acc[t][r];
        if (bias) v += bias[col];
        if (relu) v = fmaxf(v, 0.f);
        out[(long)row * ldc + col] = v;
      }
    }
  }
}

// ---------------- LayerNorm over 64 features, optional +bias / relu before ----
__global__ __launch_bounds__(256)
void ln64_kernel(const float* __restrict__ in, long ldi,
                 const float* __restrict__ addb, int relu,
                 const float* __restrict__ g, const float* __restrict__ b,
                 float* __restrict__ out, long ldo, int M)
{
  int row = blockIdx.x * 8 + (threadIdx.x >> 5);
  int lane = threadIdx.x & 31;
  if (row >= M) return;
  float v0 = in[(long)row * ldi + lane];
  float v1 = in[(long)row * ldi + lane + 32];
  if (addb) { v0 += addb[lane]; v1 += addb[lane + 32]; }
  if (relu) { v0 = fmaxf(v0, 0.f); v1 = fmaxf(v1, 0.f); }
  float s = v0 + v1;
  for (int o = 16; o; o >>= 1) s += __shfl_xor(s, o, 32);
  float mean = s * (1.f / 64.f);
  float d0 = v0 - mean, d1 = v1 - mean;
  float q = d0 * d0 + d1 * d1;
  for (int o = 16; o; o >>= 1) q += __shfl_xor(q, o, 32);
  float rstd = rsqrtf(q * (1.f / 64.f) + 1e-5f);
  out[(long)row * ldo + lane]      = d0 * rstd * g[lane] + b[lane];
  out[(long)row * ldo + lane + 32] = d1 * rstd * g[lane + 32] + b[lane + 32];
}

// ---------------- row dot: out[r] = in1[r,:]·w1 (+ in2[r,:]·w2) (+ bias) --------
__global__ __launch_bounds__(256)
void rowdot_kernel(const float* __restrict__ in1, long ld1, const float* __restrict__ w1,
                   const float* __restrict__ in2, long ld2, const float* __restrict__ w2,
                   const float* __restrict__ bias, float* __restrict__ out, int M)
{
  int row = blockIdx.x * 8 + (threadIdx.x >> 5);
  int lane = threadIdx.x & 31;
  if (row >= M) return;
  float s = in1[(long)row * ld1 + lane] * w1[lane]
          + in1[(long)row * ld1 + lane + 32] * w1[lane + 32];
  if (in2) s += in2[(long)row * ld2 + lane] * w2[lane]
              + in2[(long)row * ld2 + lane + 32] * w2[lane + 32];
  for (int o = 16; o; o >>= 1) s += __shfl_xor(s, o, 32);
  if (lane == 0) out[row] = s + (bias ? bias[0] : 0.f);
}

// ---------------- misc elementwise / scatter kernels ----------------
__global__ void zero_kernel(float* p, long n) {
  long i = (long)blockIdx.x * blockDim.x + threadIdx.x;
  for (; i < n; i += (long)gridDim.x * blockDim.x) p[i] = 0.f;
}
__global__ void zero2d_kernel(float* p, long ld, long M) {   // zero 64 cols
  long i = (long)blockIdx.x * blockDim.x + threadIdx.x;
  long n = M * 64;
  for (; i < n; i += (long)gridDim.x * blockDim.x)
    p[(i >> 6) * ld + (i & 63)] = 0.f;
}
// convert f32 [K,N] weight -> f16 in fragment-chunk (swizzled) layout
__global__ void cvt_swz_kernel(const float* __restrict__ in, _Float16* __restrict__ out,
                               int K, int N) {
  long i = (long)blockIdx.x * blockDim.x + threadIdx.x;
  long n = (long)K * N;
  for (; i < n; i += (long)gridDim.x * blockDim.x) {
    int gk = (int)(i / N), c = (int)(i - (long)gk * N);
    out[wswz(gk, c, N)] = (_Float16)in[i];
  }
}
// stacked+transposed LSTM weights [wih^T ; whh^T] -> swizzled f16 [128,256]
__global__ void lstm_stack_kernel(const float* __restrict__ wih, const float* __restrict__ whh,
                                  _Float16* __restrict__ out) {
  int i = blockIdx.x * blockDim.x + threadIdx.x;
  if (i >= 128 * 256) return;
  int gk = i >> 8, n = i & 255;
  float v = (gk < 64) ? wih[n * 64 + gk] : whh[n * 64 + (gk - 64)];
  out[wswz(gk, n, 256)] = (_Float16)v;
}
__global__ void addvec_kernel(const float* a, const float* b, float* out, int n) {
  int i = blockIdx.x * blockDim.x + threadIdx.x;
  if (i < n) out[i] = a[i] + b[i];
}
__global__ void count_kernel(const int* idx, float* cnt, int n) {
  int i = blockIdx.x * blockDim.x + threadIdx.x;
  if (i < n) atomicAdd(&cnt[idx[i]], 1.f);
}
__global__ void scatter_add_kernel(float* __restrict__ dst, long ldD,
                                   const float* __restrict__ src, long ldS,
                                   const int* __restrict__ idx, long M) {
  long i = (long)blockIdx.x * blockDim.x + threadIdx.x;
  long n = M * 64;
  for (; i < n; i += (long)gridDim.x * blockDim.x) {
    long r = i >> 6; int c = (int)(i & 63);
    atomicAdd(&dst[(long)idx[r] * ldD + c], src[r * ldS + c]);
  }
}
__global__ void meandiv_kernel(float* sums, const float* cnt) {  // [64,64]
  int i = blockIdx.x * blockDim.x + threadIdx.x;
  if (i < 64 * 64) sums[i] /= fmaxf(cnt[i >> 6], 1.f);
}

__device__ __forceinline__ unsigned fenc(float x) {
  unsigned u = __float_as_uint(x);
  return (u & 0x80000000u) ? ~u : (u | 0x80000000u);
}
__device__ __forceinline__ float fdec(unsigned u) {
  return (u & 0x80000000u) ? __uint_as_float(u & 0x7FFFFFFFu) : __uint_as_float(~u);
}
__global__ void logit_kernel(const float* __restrict__ as_, const float* __restrict__ ad_,
                             const float* __restrict__ ae_, const int* __restrict__ src,
                             const int* __restrict__ dst, float* __restrict__ logit,
                             unsigned* __restrict__ menc, int ne) {
  int i = blockIdx.x * blockDim.x + threadIdx.x;
  if (i >= ne) return;
  float l = as_[src[i]] + ad_[dst[i]] + ae_[i];
  l = (l > 0.f) ? l : 0.2f * l;                      // leaky_relu 0.2
  logit[i] = l;
  atomicMax(&menc[dst[i]], fenc(l));
}
__global__ void exden_kernel(const float* __restrict__ logit, const unsigned* __restrict__ menc,
                             const int* __restrict__ dst, float* __restrict__ exv,
                             float* __restrict__ den, int ne) {
  int i = blockIdx.x * blockDim.x + threadIdx.x;
  if (i >= ne) return;
  float e = __expf(logit[i] - fdec(menc[dst[i]]));
  exv[i] = e;
  atomicAdd(&den[dst[i]], e);
}
__global__ void gat_scatter_kernel(const float* __restrict__ exv, const float* __restrict__ den,
                                   const float* __restrict__ xs, const int* __restrict__ src,
                                   const int* __restrict__ dst, float* __restrict__ gat, long ne) {
  long i = (long)blockIdx.x * blockDim.x + threadIdx.x;
  long n = ne * 64;
  for (; i < n; i += (long)gridDim.x * blockDim.x) {
    long e = i >> 6; int c = (int)(i & 63);
    int d = dst[e];
    float alpha = exv[e] / (den[d] + 1e-16f);
    atomicAdd(&gat[(long)d * 64 + c], alpha * xs[(long)src[e] * 64 + c]);
  }
}
__device__ __forceinline__ float sigmoidf_(float x) { return 1.f / (1.f + __expf(-x)); }
__global__ void lstm_cell_kernel(const float* __restrict__ gates, const float* __restrict__ cprev,
                                 float* __restrict__ hout, float* __restrict__ cout, long M) {
  long i = (long)blockIdx.x * blockDim.x + threadIdx.x;
  long n = M * 64;
  for (; i < n; i += (long)gridDim.x * blockDim.x) {
    long r = i >> 6; int c = (int)(i & 63);
    const float* g = gates + r * 256;
    float ig = sigmoidf_(g[c]);
    float fg = sigmoidf_(g[64 + c]);
    float gg = tanhf(g[128 + c]);
    float og = sigmoidf_(g[192 + c]);
    float cp = cprev ? cprev[i] : 0.f;
    float cn = fg * cp + ig * gg;
    cout[i] = cn;
    hout[i] = og * tanhf(cn);
  }
}
__global__ void jk_combine_kernel(const float* __restrict__ s0, const float* __restrict__ s1,
                                  const float* __restrict__ mlpx, const float* __restrict__ gat,
                                  float* __restrict__ out, long ldo, long M) {
  long i = (long)blockIdx.x * blockDim.x + threadIdx.x;
  long n = M * 64;
  for (; i < n; i += (long)gridDim.x * blockDim.x) {
    long r = i >> 6; int c = (int)(i & 63);
    float v0 = s0[r], v1 = s1[r];
    float mx = fmaxf(v0, v1);
    float e0 = __expf(v0 - mx), e1 = __expf(v1 - mx);
    float a0 = e0 / (e0 + e1);
    out[r * ldo + c] = a0 * mlpx[i] + (1.f - a0) * gat[i];
  }
}
__global__ void proj_kernel(const float* __restrict__ in, long ldi,
                            const float* __restrict__ w, const float* __restrict__ b,
                            float* __restrict__ out, long M, int Nout) {
  long i = (long)blockIdx.x * blockDim.x + threadIdx.x;
  long n = M * Nout;
  for (; i < n; i += (long)gridDim.x * blockDim.x) {
    long r = i / Nout; int c = (int)(i - r * Nout);
    float s = b[c];
    #pragma unroll
    for (int k = 0; k < 64; ++k) s += in[r * ldi + k] * w[k * Nout + c];
    out[r * Nout + c] = s;
  }
}

// ---------------- host-side helpers ----------------
static inline GemmA segA(const float* b0, const int* i0, int k0, int s0,
                         const float* b1 = nullptr, const int* i1 = nullptr, int k1 = 0, int s1 = 0,
                         const float* b2 = nullptr, const int* i2 = nullptr, int k2 = 0, int s2 = 0,
                         const float* b3 = nullptr, const int* i3 = nullptr, int k3 = 0, int s3 = 0) {
  GemmA A;
  A.base0 = b0; A.base1 = b1; A.base2 = b2; A.base3 = b3;
  A.idx0 = i0;  A.idx1 = i1;  A.idx2 = i2;  A.idx3 = i3;
  A.kw0 = k0;   A.kw1 = k1;   A.kw2 = k2;   A.kw3 = k3;
  A.st0 = s0;   A.st1 = s1;   A.st2 = s2;   A.st3 = s3;
  return A;
}

extern "C" void kernel_launch(void* const* d_in, const int* in_sizes, int n_in,
                              void* d_out, int out_size, void* d_ws, size_t ws_size,
                              hipStream_t stream) {
  if (n_in < 84) return;
  (void)in_sizes; (void)out_size; (void)ws_size;

  const float* F[84];
  for (int i = 0; i < 84; ++i) F[i] = (const float*)d_in[i];
  const int* src   = (const int*)d_in[IN_EIDX];
  const int* dst   = src + NEN;
  const int* v_idx = (const int*)d_in[IN_VIDX];
  const int* e_idx = (const int*)d_in[IN_EINDICES];
  float* outp = (float*)d_out;

  // ---- workspace bump allocator ----
  char* wsp = (char*)d_ws; size_t wso = 0;
  auto allocB = [&](size_t bytes) -> void* {
    void* p = wsp + wso; wso = (wso + bytes + 255) & ~(size_t)255; return p;
  };
  auto aF = [&](size_t n) { return (float*)allocB(n * 4); };
  auto aH = [&](size_t n) { return (_Float16*)allocB(n * 2); };

  // f16 weights (swizzled; K padded to multiple of 32)
  _Float16 *hCE1 = aH(512 * 128), *hCE2 = aH(128 * 64);
  _Float16 *hCV1 = aH(320 * 128), *hCV2 = aH(128 * 64);
  _Float16 *hCU1 = aH(256 * 128), *hCU2 = aH(128 * 64);
  _Float16 *hEV1 = aH(32 * 128),  *hEV2 = aH(128 * 64);   // K=8  -> pad 32
  _Float16 *hEE1 = aH(32 * 128),  *hEE2 = aH(128 * 64);   // K=4  -> pad 32
  _Float16 *hEU1 = aH(32 * 128),  *hEU2 = aH(128 * 64);   // K=4  -> pad 32
  _Float16 *hDV1 = aH(64 * 128),  *hDV2 = aH(128 * 64);
  _Float16 *hDE1 = aH(64 * 128),  *hDE2 = aH(128 * 64);
  _Float16 *hDU1 = aH(64 * 128),  *hDU2 = aH(128 * 64);
  _Float16 *hLIN = aH(64 * 64),   *hLINE = aH(64 * 64);
  _Float16 *hLF  = aH(128 * 256), *hLB   = aH(128 * 256);
  float *bF = aF(256), *bB = aF(256);

  // activations
  float* x2  = aF((size_t)NVN * 128);   // [x0 | hx]
  float* e2  = aF((size_t)NEN * 128);   // [e0 | he]
  float* u2  = aF((size_t)NUN * 128);   // [u0 | hu]
  float* tH  = aF((size_t)NEN * 128);   // hidden-stage / gates temp
  float* tA  = aF((size_t)NEN * 64);
  float* tB  = aF((size_t)NEN * 64);
  float* mlpx = aF((size_t)NVN * 64);
  float* gat  = aF((size_t)NVN * 64);
  float* agg  = aF((size_t)NVN * 64);
  float* hf1 = aF((size_t)NVN * 64); float* cf1 = aF((size_t)NVN * 64);
  float* hf2 = aF((size_t)NVN * 64);
  float* hb1 = aF((size_t)NVN * 64); float* cb1 = aF((size_t)NVN * 64);
  float* hb2 = aF((size_t)NVN * 64);
  float* as_ = aF(NVN); float* ad_ = aF(NVN);
  float* ae_ = aF(NEN); float* logit = aF(NEN); float* exv = aF(NEN);
  unsigned* menc = (unsigned*)allocB(NVN * 4);
  float* den = aF(NVN); float* s0v = aF(NVN); float* s1v = aF(NVN);
  float* meanx = aF(64 * 64); float* meane = aF(64 * 64);
  float* cntV = aF(64); float* cntE = aF(64);

  // ---- launch helpers ----
  auto gemm = [&](const GemmA& A, const _Float16* W, const float* bias, float* out,
                  int ldc, int M, int N, int K, int relu) {
    dim3 grid((M + 31) / 32);
    if (N == 64)
      gemm_wmma_kernel<64><<<grid, dim3(256), 0, stream>>>(A, W, bias, out, (long)ldc, M, K, relu);
    else if (N == 128)
      gemm_wmma_kernel<128><<<grid, dim3(256), 0, stream>>>(A, W, bias, out, (long)ldc, M, K, relu);
    else
      gemm_wmma_kernel<256><<<grid, dim3(256), 0, stream>>>(A, W, bias, out, (long)ldc, M, K, relu);
  };
  auto ln = [&](const float* in, int ldi, const float* addb, int relu,
                const float* g, const float* b, float* out, int ldo, int M) {
    ln64_kernel<<<dim3((M + 7) / 8), dim3(256), 0, stream>>>(
        in, (long)ldi, addb, relu, g, b, out, (long)ldo, M);
  };
  auto rowdot = [&](const float* i1, int l1, const float* w1, const float* i2, int l2,
                    const float* w2, const float* bias, float* out, int M) {
    rowdot_kernel<<<dim3((M + 7) / 8), dim3(256), 0, stream>>>(
        i1, (long)l1, w1, i2, (long)l2, w2, bias, out, M);
  };
  auto zero = [&](float* p, long n) {
    zero_kernel<<<dim3((int)((n + 255) / 256) > 65535 ? 65535 : (int)((n + 255) / 256)),
                  dim3(256), 0, stream>>>(p, n);
  };
  auto cvtw = [&](const float* in, _Float16* out, int K, int N) {
    long n = (long)K * N;
    cvt_swz_kernel<<<dim3((int)((n + 255) / 256)), dim3(256), 0, stream>>>(in, out, K, N);
  };
  auto ew = [&](long n) { long g = (n + 255) / 256; return dim3((int)(g > 65535 ? 65535 : g)); };

  // ---- weight prep: f32 -> f16 swizzled (LSTM transposed+stacked) ----
  zero((float*)hEV1, 32 * 128 / 2);   // zero K-pad regions (halves as floats)
  zero((float*)hEE1, 32 * 128 / 2);
  zero((float*)hEU1, 32 * 128 / 2);
  cvtw(F[P_CE_W0], hCE1, 512, 128); cvtw(F[P_CE_W1], hCE2, 128, 64);
  cvtw(F[P_CV_W0], hCV1, 320, 128); cvtw(F[P_CV_W1], hCV2, 128, 64);
  cvtw(F[P_CU_W0], hCU1, 256, 128); cvtw(F[P_CU_W1], hCU2, 128, 64);
  cvtw(F[P_EV_W0], hEV1, 8, 128);   cvtw(F[P_EV_W1], hEV2, 128, 64);
  cvtw(F[P_EE_W0], hEE1, 4, 128);   cvtw(F[P_EE_W1], hEE2, 128, 64);
  cvtw(F[P_EU_W0], hEU1, 4, 128);   cvtw(F[P_EU_W1], hEU2, 128, 64);
  cvtw(F[P_DV_W0], hDV1, 64, 128);  cvtw(F[P_DV_W1], hDV2, 128, 64);
  cvtw(F[P_DE_W0], hDE1, 64, 128);  cvtw(F[P_DE_W1], hDE2, 128, 64);
  cvtw(F[P_DU_W0], hDU1, 64, 128);  cvtw(F[P_DU_W1], hDU2, 128, 64);
  cvtw(F[P_LIN],  hLIN,  64, 64);   cvtw(F[P_LIN_E], hLINE, 64, 64);
  lstm_stack_kernel<<<dim3(128), dim3(256), 0, stream>>>(F[P_LSTMF_WIH], F[P_LSTMF_WHH], hLF);
  lstm_stack_kernel<<<dim3(128), dim3(256), 0, stream>>>(F[P_LSTMB_WIH], F[P_LSTMB_WHH], hLB);
  addvec_kernel<<<dim3(1), dim3(256), 0, stream>>>(F[P_LSTMF_BIH], F[P_LSTMF_BHH], bF, 256);
  addvec_kernel<<<dim3(1), dim3(256), 0, stream>>>(F[P_LSTMB_BIH], F[P_LSTMB_BHH], bB, 256);

  // ---- segment counts (reused both steps) ----
  zero(cntV, 64); zero(cntE, 64);
  count_kernel<<<dim3((NVN + 255) / 256), dim3(256), 0, stream>>>(v_idx, cntV, NVN);
  count_kernel<<<dim3((NEN + 255) / 256), dim3(256), 0, stream>>>(e_idx, cntE, NEN);

  // ================= Encoder =================
  gemm(segA(F[IN_X], nullptr, 8, 8), hEV1, F[P_EV_B0], tH, 128, NVN, 128, 8, 1);
  gemm(segA(tH, nullptr, 128, 128), hEV2, F[P_EV_B1], tA, 64, NVN, 64, 128, 1);
  ln(tA, 64, nullptr, 0, F[P_EV_LNG], F[P_EV_LNB], x2, 128, NVN);
  zero2d_kernel<<<ew((long)NVN * 64), dim3(256), 0, stream>>>(x2 + 64, 128, NVN);

  gemm(segA(F[IN_EA], nullptr, 4, 4), hEE1, F[P_EE_B0], tH, 128, NEN, 128, 4, 1);
  gemm(segA(tH, nullptr, 128, 128), hEE2, F[P_EE_B1], tA, 64, NEN, 64, 128, 1);
  ln(tA, 64, nullptr, 0, F[P_EE_LNG], F[P_EE_LNB], e2, 128, NEN);
  zero2d_kernel<<<ew((long)NEN * 64), dim3(256), 0, stream>>>(e2 + 64, 128, NEN);

  gemm(segA(F[IN_U], nullptr, 4, 4), hEU1, F[P_EU_B0], tH, 128, NUN, 128, 4, 1);
  gemm(segA(tH, nullptr, 128, 128), hEU2, F[P_EU_B1], tA, 64, NUN, 64, 128, 1);
  ln(tA, 64, nullptr, 0, F[P_EU_LNG], F[P_EU_LNB], u2, 128, NUN);
  zero2d_kernel<<<ew((long)NUN * 64), dim3(256), 0, stream>>>(u2 + 64, 128, NUN);

  // ================= Core (2 steps) =================
  for (int step = 0; step < 2; ++step) {
    // ---- edge block: concat(x[src], x[dst], e, u[e_idx]) [E,512] -> MLP -> LN ----
    gemm(segA(x2, src, 128, 128, x2, dst, 128, 128, e2, nullptr, 128, 128, u2, e_idx, 128, 128),
         hCE1, F[P_CE_B0], tH, 128, NEN, 128, 512, 1);
    gemm(segA(tH, nullptr, 128, 128), hCE2, F[P_CE_B1], tA, 64, NEN, 64, 128, 1);
    ln(tA, 64, nullptr, 0, F[P_CE_LNG], F[P_CE_LNB], e2 + 64, 128, NEN);   // e_new

    // ---- node aggregation: segment_sum(e_new over src) ----
    zero(agg, (long)NVN * 64);
    scatter_add_kernel<<<ew((long)NEN * 64), dim3(256), 0, stream>>>(agg, 64, e2 + 64, 128, src, NEN);

    // ---- node block: concat(x, agg, u[v_idx]) [V,320] -> MLP -> LN ----
    gemm(segA(x2, nullptr, 128, 128, agg, nullptr, 64, 64, u2, v_idx, 128, 128),
         hCV1, F[P_CV_B0], tH, 128, NVN, 128, 320, 1);
    gemm(segA(tH, nullptr, 128, 128), hCV2, F[P_CV_B1], tA, 64, NVN, 64, 128, 1);
    ln(tA, 64, nullptr, 0, F[P_CV_LNG], F[P_CV_LNB], mlpx, 64, NVN);

    // ---- GAT attention ----
    gemm(segA(mlpx, nullptr, 64, 64), hLIN, nullptr, tA, 64, NVN, 64, 64, 0);       // xs
    gemm(segA(e2 + 64, nullptr, 64, 128), hLINE, nullptr, tB, 64, NEN, 64, 64, 0);  // ee
    rowdot(tA, 64, F[P_A_SRC], nullptr, 0, nullptr, nullptr, as_, NVN);
    rowdot(tA, 64, F[P_A_DST], nullptr, 0, nullptr, nullptr, ad_, NVN);
    rowdot(tB, 64, F[P_A_EDGE], nullptr, 0, nullptr, nullptr, ae_, NEN);
    zero((float*)menc, NVN); zero(den, NVN);
    logit_kernel<<<dim3((NEN + 255) / 256), dim3(256), 0, stream>>>(as_, ad_, ae_, src, dst, logit, menc, NEN);
    exden_kernel<<<dim3((NEN + 255) / 256), dim3(256), 0, stream>>>(logit, menc, dst, exv, den, NEN);
    zero(gat, (long)NVN * 64);
    gat_scatter_kernel<<<ew((long)NEN * 64), dim3(256), 0, stream>>>(exv, den, tA, src, dst, gat, NEN);
    ln(gat, 64, F[P_ATT_BIAS], 1, F[P_ATT_LNG], F[P_ATT_LNB], gat, 64, NVN);  // relu(gat+b) -> LN

    // ---- bidirectional LSTM JumpingKnowledge over (mlpx, gat) ----
    gemm(segA(mlpx, nullptr, 64, 64), hLF, bF, tH, 256, NVN, 256, 64, 0);
    lstm_cell_kernel<<<ew((long)NVN * 64), dim3(256), 0, stream>>>(tH, nullptr, hf1, cf1, NVN);
    gemm(segA(gat, nullptr, 64, 64, hf1, nullptr, 64, 64), hLF, bF, tH, 256, NVN, 256, 128, 0);
    lstm_cell_kernel<<<ew((long)NVN * 64), dim3(256), 0, stream>>>(tH, cf1, hf2, cf1, NVN);
    gemm(segA(gat, nullptr, 64, 64), hLB, bB, tH, 256, NVN, 256, 64, 0);
    lstm_cell_kernel<<<ew((long)NVN * 64), dim3(256), 0, stream>>>(tH, nullptr, hb1, cb1, NVN);
    gemm(segA(mlpx, nullptr, 64, 64, hb1, nullptr, 64, 64), hLB, bB, tH, 256, NVN, 256, 128, 0);
    lstm_cell_kernel<<<ew((long)NVN * 64), dim3(256), 0, stream>>>(tH, cb1, hb2, cb1, NVN);
    rowdot(hf1, 64, F[P_ATT_W], hb2, 64, F[P_ATT_W] + 64, F[P_ATT_B], s0v, NVN);
    rowdot(hf2, 64, F[P_ATT_W], hb1, 64, F[P_ATT_W] + 64, F[P_ATT_B], s1v, NVN);
    jk_combine_kernel<<<ew((long)NVN * 64), dim3(256), 0, stream>>>(s0v, s1v, mlpx, gat, x2 + 64, 128, NVN); // x_new

    // ---- global block ----
    zero(meanx, 64 * 64); zero(meane, 64 * 64);
    scatter_add_kernel<<<ew((long)NVN * 64), dim3(256), 0, stream>>>(meanx, 64, x2 + 64, 128, v_idx, NVN);
    scatter_add_kernel<<<ew((long)NEN * 64), dim3(256), 0, stream>>>(meane, 64, e2 + 64, 128, e_idx, NEN);
    meandiv_kernel<<<dim3(16), dim3(256), 0, stream>>>(meanx, cntV);
    meandiv_kernel<<<dim3(16), dim3(256), 0, stream>>>(meane, cntE);
    gemm(segA(u2, nullptr, 128, 128, meanx, nullptr, 64, 64, meane, nullptr, 64, 64),
         hCU1, F[P_CU_B0], tH, 128, NUN, 128, 256, 1);
    gemm(segA(tH, nullptr, 128, 128), hCU2, F[P_CU_B1], tA, 64, NUN, 64, 128, 1);
    ln(tA, 64, nullptr, 0, F[P_CU_LNG], F[P_CU_LNB], u2 + 64, 128, NUN);    // u_new
  }

  // ================= Decoder + output heads =================
  // nodes
  gemm(segA(x2 + 64, nullptr, 64, 128), hDV1, F[P_DV_B0], tH, 128, NVN, 128, 64, 1);
  gemm(segA(tH, nullptr, 128, 128), hDV2, F[P_DV_B1], tA, 64, NVN, 64, 128, 1);
  ln(tA, 64, nullptr, 0, F[P_DV_LNG], F[P_DV_LNB], tA, 64, NVN);
  proj_kernel<<<ew((long)NVN * 2), dim3(256), 0, stream>>>(tA, 64, F[P_OV_W], F[P_OV_B], outp, NVN, 2);
  // edges
  gemm(segA(e2 + 64, nullptr, 64, 128), hDE1, F[P_DE_B0], tH, 128, NEN, 128, 64, 1);
  gemm(segA(tH, nullptr, 128, 128), hDE2, F[P_DE_B1], tA, 64, NEN, 64, 128, 1);
  ln(tA, 64, nullptr, 0, F[P_DE_LNG], F[P_DE_LNB], tA, 64, NEN);
  proj_kernel<<<ew((long)NEN * 2), dim3(256), 0, stream>>>(tA, 64, F[P_OE_W], F[P_OE_B], outp + (long)NVN * 2, NEN, 2);
  // globals
  gemm(segA(u2 + 64, nullptr, 64, 128), hDU1, F[P_DU_B0], tH, 128, NUN, 128, 64, 1);
  gemm(segA(tH, nullptr, 128, 128), hDU2, F[P_DU_B1], tA, 64, NUN, 64, 128, 1);
  ln(tA, 64, nullptr, 0, F[P_DU_LNG], F[P_DU_LNB], tA, 64, NUN);
  proj_kernel<<<dim3(1), dim3(256), 0, stream>>>(tA, 64, F[P_OU_W], F[P_OU_B], outp + (long)NVN * 2 + (long)NEN * 2, NUN, 1);
}
